// TrivialDecoder_1236950581455
// MI455X (gfx1250) — matrix operand order repaired and checked by reading, not compiled, not run on hardware
//
#include <hip/hip_runtime.h>
#include <hip/hip_bf16.h>

typedef _Float16 v16h __attribute__((ext_vector_type(16)));
typedef _Float16 v8h  __attribute__((ext_vector_type(8)));
typedef float    v8f  __attribute__((ext_vector_type(8)));

#define THREADS 256
#define PAIRS_PER_BLOCK 64

// Load 8 consecutive f16 feature values for feature row `row`, feature index q (multiple of 8).
// Feature layout: k in [0,64) -> u[k]; [64,128) -> v[k-64]; [128,192) -> u[k-128]*v[k-128].
// sUV row layout: 128 f16 per pair = u[0..63] | v[0..63].
__device__ __forceinline__ v8h load_feat(const _Float16* sUV, int row, int q) {
    if (q < 128) {
        return *(const v8h*)(sUV + row * 128 + q);
    } else {
        int j = q - 128;
        v8h u = *(const v8h*)(sUV + row * 128 + j);
        v8h v = *(const v8h*)(sUV + row * 128 + 64 + j);
        return u * v;   // packed f16 multiplies
    }
}

__global__ __launch_bounds__(THREADS)
void mlp_score_kernel(const float* __restrict__ embeds,      // [N][64] f32
                      const int*   __restrict__ pairs,       // [count][2]
                      const float* __restrict__ W1,          // [64][192] row-major
                      const float* __restrict__ b1,          // [64]
                      const float* __restrict__ W2,          // [64]
                      const float* __restrict__ b2,          // [1]
                      float*       __restrict__ scores,      // [count]
                      long long count)
{
    __shared__ __align__(16) _Float16 sUV[PAIRS_PER_BLOCK * 128];  // 16 KB
    __shared__ __align__(16) _Float16 sW1[64 * 192];               // 24 KB
    __shared__ float sH[PAIRS_PER_BLOCK * 65];                     // 16.25 KB (padded stride)
    __shared__ float sW2[64];
    __shared__ float sB1[64];
    __shared__ int   sIU[PAIRS_PER_BLOCK];
    __shared__ int   sIV[PAIRS_PER_BLOCK];

    const int tid = threadIdx.x;
    const long long base = (long long)blockIdx.x * PAIRS_PER_BLOCK;

    // ---- Stage weights + pair indices into LDS ----
    for (int i = tid; i < 64 * 192; i += THREADS) sW1[i] = (_Float16)W1[i];
    if (tid < 64) { sW2[tid] = W2[tid]; sB1[tid] = b1[tid]; }
    if (tid < PAIRS_PER_BLOCK) {
        long long p = base + tid;
        int iu = 0, iv = 0;
        if (p < count) { iu = pairs[2 * p]; iv = pairs[2 * p + 1]; }
        sIU[tid] = iu; sIV[tid] = iv;
    }
    __syncthreads();

    // ---- Gather embeddings (coalesced over j) -> LDS f16 ----
    for (int t = tid; t < PAIRS_PER_BLOCK * 64; t += THREADS) {
        int p = t >> 6, j = t & 63;
        float uf = embeds[(long long)sIU[p] * 64 + j];
        float vf = embeds[(long long)sIV[p] * 64 + j];
        sUV[p * 128 + j]      = (_Float16)uf;
        sUV[p * 128 + 64 + j] = (_Float16)vf;
    }
    __syncthreads();

    // ---- Layer 1 GEMM via WMMA: h(64x64) = X(64x192) * W1^T(192x64) ----
    const int wave  = tid >> 5;
    const int lane  = tid & 31;
    const int l16   = lane & 15;
    const int hi    = (lane & 16) ? 1 : 0;
    const int mtile = wave & 3;                 // pairs [16*mtile, 16*mtile+16)
    const int nlo   = (wave >> 2) * 2;          // this wave does N-tiles nlo, nlo+1
    const int arow  = mtile * 16 + l16;         // A-matrix row this lane supplies

    for (int nn = 0; nn < 2; ++nn) {
        const int n   = nlo + nn;
        const int col = n * 16 + l16;           // output channel for this lane
        const float bias = sB1[col];
        v8f acc;
        #pragma unroll
        for (int j = 0; j < 8; ++j) acc[j] = bias;

        #pragma unroll
        for (int kc = 0; kc < 6; ++kc) {
            // A fragment: lane<16 holds K {0..7,16..23} of the chunk, lane>=16 holds {8..15,24..31}
            const int q0 = kc * 32 + hi * 8;
            v8h alo = load_feat(sUV, arow, q0);
            v8h ahi = load_feat(sUV, arow, q0 + 16);
            // B fragment: lane holds one column (output channel), K {0..7,16..23} (lane<16) or {8..15,24..31} (lane>=16)
            const int kb = kc * 32 + hi * 8;
            v8h blo = *(const v8h*)(sW1 + col * 192 + kb);
            v8h bhi = *(const v8h*)(sW1 + col * 192 + kb + 16);
            v16h a, b;
            #pragma unroll
            for (int i = 0; i < 8; ++i) {
                a[i] = alo[i]; a[i + 8] = ahi[i];
                b[i] = blo[i]; b[i + 8] = bhi[i];
            }
            acc = __builtin_amdgcn_wmma_f32_16x16x32_f16(
                      false, a, false, b, (short)0, acc, false, false);
        }

        // ReLU + spill D to LDS (row stride 65 -> conflict-free column access)
        #pragma unroll
        for (int j = 0; j < 8; ++j) {
            int m = mtile * 16 + j + hi * 8;
            float hv = acc[j];
            sH[m * 65 + col] = hv > 0.0f ? hv : 0.0f;
        }
    }
    __syncthreads();

    // ---- Layer 2: per-pair 64-dot, sigmoid, exp ----
    if (tid < PAIRS_PER_BLOCK) {
        long long p = base + tid;
        float s = b2[0];
        #pragma unroll
        for (int o = 0; o < 64; ++o) s += sH[tid * 65 + o] * sW2[o];
        float sig = 1.0f / (1.0f + expf(-s));
        float scr = expf(sig);
        if (p < count) scores[p] = scr;
    }
}

__global__ __launch_bounds__(THREADS)
void group_reduce_kernel(const float* __restrict__ pos_scr,
                         const float* __restrict__ neg_scr,
                         float* __restrict__ partials,
                         int E, int K)
{
    __shared__ float red[THREADS];
    int e = blockIdx.x * THREADS + threadIdx.x;
    float val = 0.0f;
    if (e < E) {
        float ps = pos_scr[e];
        float denom = ps;
        for (int k = 0; k < K; ++k) denom += neg_scr[(long long)e * K + k];
        val = ps / (denom + 1e-8f) + 1e-8f;
    }
    red[threadIdx.x] = val;
    __syncthreads();
    for (int s = THREADS / 2; s > 0; s >>= 1) {
        if (threadIdx.x < s) red[threadIdx.x] += red[threadIdx.x + s];
        __syncthreads();
    }
    if (threadIdx.x == 0) partials[blockIdx.x] = red[0];
}

__global__ __launch_bounds__(THREADS)
void final_reduce_kernel(const float* __restrict__ partials, int n,
                         float* __restrict__ out)
{
    __shared__ float red[THREADS];
    float v = 0.0f;
    for (int i = threadIdx.x; i < n; i += THREADS) v += partials[i];
    red[threadIdx.x] = v;
    __syncthreads();
    for (int s = THREADS / 2; s > 0; s >>= 1) {
        if (threadIdx.x < s) red[threadIdx.x] += red[threadIdx.x + s];
        __syncthreads();
    }
    if (threadIdx.x == 0) out[0] = -red[0];
}

extern "C" void kernel_launch(void* const* d_in, const int* in_sizes, int n_in,
                              void* d_out, int out_size, void* d_ws, size_t ws_size,
                              hipStream_t stream) {
    const float* embeds = (const float*)d_in[0];   // [N][64]
    const int*   pos    = (const int*)d_in[1];     // [E][2]
    const int*   neg    = (const int*)d_in[2];     // [E][K][2]
    const float* W1     = (const float*)d_in[3];   // [64][192]
    const float* b1     = (const float*)d_in[4];   // [64]
    const float* W2     = (const float*)d_in[5];   // [1][64]
    const float* b2     = (const float*)d_in[6];   // [1]

    const long long E  = in_sizes[1] / 2;          // 65536
    const long long NK = in_sizes[2] / 2;          // E*K = 1048576
    const int K = (int)(NK / E);                   // 16

    float* ws       = (float*)d_ws;
    float* neg_scr  = ws;                          // NK floats
    float* pos_scr  = ws + NK;                     // E floats
    float* partials = pos_scr + E;                 // reduce blocks

    const int negBlocks = (int)((NK + PAIRS_PER_BLOCK - 1) / PAIRS_PER_BLOCK);
    const int posBlocks = (int)((E  + PAIRS_PER_BLOCK - 1) / PAIRS_PER_BLOCK);

    mlp_score_kernel<<<negBlocks, THREADS, 0, stream>>>(
        embeds, neg, W1, b1, W2, b2, neg_scr, NK);
    mlp_score_kernel<<<posBlocks, THREADS, 0, stream>>>(
        embeds, pos, W1, b1, W2, b2, pos_scr, E);

    const int rBlocks = (int)((E + THREADS - 1) / THREADS);
    group_reduce_kernel<<<rBlocks, THREADS, 0, stream>>>(
        pos_scr, neg_scr, partials, (int)E, K);
    final_reduce_kernel<<<1, THREADS, 0, stream>>>(
        partials, rBlocks, (float*)d_out);
}
